// SingleHeadAttention_6485400617233
// MI455X (gfx1250) — matrix operand order repaired and checked by reading, not compiled, not run on hardware
//
#include <hip/hip_runtime.h>

typedef __attribute__((ext_vector_type(16))) __bf16 v16bf;
typedef __attribute__((ext_vector_type(8)))  __bf16 v8bf;
typedef __attribute__((ext_vector_type(8)))  float  v8f;

#define SLEN 4096
#define DDIM 1024
#define BATCH 4

// ---------------------------------------------------------------------------
// WMMA helpers (gfx1250, wave32). D = A(16x32 bf16) * B(32x16 bf16) + C(f32)
// ---------------------------------------------------------------------------
__device__ __forceinline__ v8f wmma_bf16(v16bf a, v16bf b, v8f c) {
  // (neg_a, A, neg_b, B, c_mod, C, reuse_a, reuse_b)
  return __builtin_amdgcn_wmma_f32_16x16x32_bf16(false, a, false, b, (short)0, c,
                                                 false, false);
}

__device__ __forceinline__ v16bf make_frag(v8bf lo, v8bf hi) {
  v16bf r;
#pragma unroll
  for (int i = 0; i < 8; ++i) { r[i] = lo[i]; r[i + 8] = hi[i]; }
  return r;
}

// A fragment 16x32: row-major source, row stride ld (elements), K contiguous.
// lane: m = lane&15; kOff = (lane<16)?0:8; elems 0..7 -> k=kOff+i, 8..15 -> k=kOff+16+i
__device__ __forceinline__ v16bf load_a_frag(const __bf16* base, int ld, int lane) {
  const __bf16* p = base + (lane & 15) * ld + ((lane & 16) ? 8 : 0);
  v8bf lo = *(const v8bf*)p;
  v8bf hi = *(const v8bf*)(p + 16);
  return make_frag(lo, hi);
}

// B fragment 32x16: source laid out [n][k] row-major (contraction contiguous),
// row stride ld. lane: n = lane&15; kOff = (lane<16)?0:8; elems 0..7 -> k=kOff+i, 8..15 -> k=kOff+16+i
__device__ __forceinline__ v16bf load_b_frag(const __bf16* base, int ld, int lane) {
  const __bf16* p = base + (lane & 15) * ld + ((lane & 16) ? 8 : 0);
  v8bf lo = *(const v8bf*)p;
  v8bf hi = *(const v8bf*)(p + 16);
  return make_frag(lo, hi);
}

// ---------------------------------------------------------------------------
// Kernel 0: convert x (f32) -> xb (bf16), vectorized 8-wide
// ---------------------------------------------------------------------------
__global__ __launch_bounds__(256) void xconv_kernel(const float* __restrict__ x,
                                                    __bf16* __restrict__ xb) {
  const size_t i = ((size_t)blockIdx.x * 256 + threadIdx.x) * 8;
  v8f v = *(const v8f*)(x + i);
  v8bf o;
#pragma unroll
  for (int j = 0; j < 8; ++j) o[j] = (__bf16)v[j];
  *(v8bf*)(xb + i) = o;
}

// ---------------------------------------------------------------------------
// Kernel 1: transpose + convert weights to bf16: Wt[n][k] = (bf16)W[k][n]
// ---------------------------------------------------------------------------
__global__ __launch_bounds__(256) void wtrans_kernel(
    const float* __restrict__ Wq, const float* __restrict__ Wk,
    const float* __restrict__ Wv,
    __bf16* __restrict__ Wqt, __bf16* __restrict__ Wkt, __bf16* __restrict__ Wvt) {
  const int which = blockIdx.z;
  const float* W = (which == 0) ? Wq : (which == 1) ? Wk : Wv;
  __bf16* Wt = (which == 0) ? Wqt : (which == 1) ? Wkt : Wvt;
  const int idx = blockIdx.x * 256 + threadIdx.x;   // 0 .. 1M-1
  const int n = idx >> 10;
  const int k = idx & 1023;
  Wt[(size_t)n * DDIM + k] = (__bf16)W[(size_t)k * DDIM + n];
}

// ---------------------------------------------------------------------------
// Kernel 2: fused QKV projection.  y = xb @ W + b  (M=16384, N=1024, K=1024)
// blockIdx.z selects q/k/v.  Q,K stored row-major bf16; V stored transposed
// (Vt[b][d][s]) so the PV GEMM gets contiguous B-fragments.
// Block: 256 thr = 8 waves -> 64x64 tile; wave = 16 rows x 32 cols (2 accs).
// ---------------------------------------------------------------------------
__global__ __launch_bounds__(256) void proj_kernel(
    const __bf16* __restrict__ xb,
    const __bf16* __restrict__ Wqt, const __bf16* __restrict__ Wkt,
    const __bf16* __restrict__ Wvt,
    const float* __restrict__ bq, const float* __restrict__ bk,
    const float* __restrict__ bv,
    __bf16* __restrict__ Qb, __bf16* __restrict__ Kb, __bf16* __restrict__ Vt) {
  const int tid = threadIdx.x;
  const int w = tid >> 5;
  const int lane = tid & 31;
  const int lh = lane & 15;
  const int rowoff = (lane & 16) ? 8 : 0;

  const int which = blockIdx.z;
  const __bf16* Wt = (which == 0) ? Wqt : (which == 1) ? Wkt : Wvt;
  const float* bias = (which == 0) ? bq : (which == 1) ? bk : bv;

  const int m0 = blockIdx.y * 64 + (w & 3) * 16;
  const int c0 = blockIdx.x * 64 + (w >> 2) * 32;

  v8f acc0 = {}, acc1 = {};
#pragma unroll 4
  for (int kc = 0; kc < DDIM / 32; ++kc) {
    v16bf af = load_a_frag(xb + (size_t)m0 * DDIM + kc * 32, DDIM, lane);
    v16bf b0 = load_b_frag(Wt + (size_t)c0 * DDIM + kc * 32, DDIM, lane);
    v16bf b1 = load_b_frag(Wt + (size_t)(c0 + 16) * DDIM + kc * 32, DDIM, lane);
    acc0 = wmma_bf16(af, b0, acc0);
    acc1 = wmma_bf16(af, b1, acc1);
  }
  const float bb0 = bias[c0 + lh];
  const float bb1 = bias[c0 + 16 + lh];

  if (which < 2) {
    __bf16* o = (which == 0) ? Qb : Kb;
#pragma unroll
    for (int r = 0; r < 8; ++r) {
      const size_t row = (size_t)(m0 + r + rowoff) * DDIM;
      o[row + c0 + lh]      = (__bf16)(acc0[r] + bb0);
      o[row + c0 + 16 + lh] = (__bf16)(acc1[r] + bb1);
    }
  } else {
#pragma unroll
    for (int r = 0; r < 8; ++r) {
      const int m = m0 + r + rowoff;
      const int bidx = m >> 12;       // batch
      const int sidx = m & 4095;      // seq position
      Vt[((size_t)bidx * DDIM + c0 + lh) * SLEN + sidx]      = (__bf16)(acc0[r] + bb0);
      Vt[((size_t)bidx * DDIM + c0 + 16 + lh) * SLEN + sidx] = (__bf16)(acc1[r] + bb1);
    }
  }
}

// ---------------------------------------------------------------------------
// Kernel 3: flash attention (causal, online softmax).
// Block = 16 query rows, 8 waves. Per key-block of 128:
//   wave w: 16x16 score tile for keys [k0+16w, k0+16w+16)  (32 WMMAs over d)
//   cross-wave softmax via LDS; P(16x128) bf16 in LDS
//   wave w: O[:, 128w:128w+128] += P @ V   (32 WMMAs)
// Q tile staged into LDS via CDNA5 async-copy (ASYNCcnt path).
// ---------------------------------------------------------------------------
__global__ __launch_bounds__(256) void attn_kernel(
    const __bf16* __restrict__ Qb, const __bf16* __restrict__ Kb,
    const __bf16* __restrict__ Vt, float* __restrict__ out) {
  __shared__ __align__(16) __bf16 Qs[16 * DDIM];   // 32 KB
  __shared__ __align__(16) __bf16 Ps[16 * 128];    // 4 KB
  __shared__ float wmaxs[8][16];
  __shared__ float wsums[8][16];
  __shared__ float rowmax[16], rowsum[16], rowscale[16];

  const int tid = threadIdx.x;
  const int w = tid >> 5;
  const int lane = tid & 31;
  const int lh = lane & 15;
  const int rowoff = (lane & 16) ? 8 : 0;

  const int b = blockIdx.y;
  const int m0 = blockIdx.x * 16;

  const __bf16* Qbase = Qb + ((size_t)b * SLEN + m0) * DDIM;
  const __bf16* Kbase = Kb + (size_t)b * SLEN * DDIM;
  const __bf16* Vbase = Vt + (size_t)b * DDIM * SLEN;

  { // stage Q tile (32 KB) into LDS with async global->LDS DMA.
    // LDS aperture maps flat addr[31:0] -> wave-relative LDS offset.
    const uint32_t lds0 = (uint32_t)(uintptr_t)(&Qs[0]);
#pragma unroll
    for (int i = 0; i < 8; ++i) {
      const uint32_t loff = lds0 + (uint32_t)(tid + i * 256) * 16u;
      const uint64_t gaddr =
          (uint64_t)(uintptr_t)((const char*)Qbase + (size_t)(tid + i * 256) * 16);
      asm volatile("global_load_async_to_lds_b128 %0, %1, off"
                   :: "v"(loff), "v"(gaddr) : "memory");
    }
    asm volatile("s_wait_asynccnt 0x0" ::: "memory");
  }
  if (tid < 16) { rowmax[tid] = -1e30f; rowsum[tid] = 0.0f; }
  __syncthreads();

  v8f acc[8];
#pragma unroll
  for (int t = 0; t < 8; ++t) acc[t] = {};

  const int nIter = (m0 + 15) / 128 + 1;
  for (int it = 0; it < nIter; ++it) {
    const int k0 = it * 128;
    const int n0 = k0 + w * 16;

    // ---- scores S = Q K^T ----
    v8f s = {};
#pragma unroll 4
    for (int kc = 0; kc < DDIM / 32; ++kc) {
      v16bf a  = load_a_frag(Qs + kc * 32, DDIM, lane);
      v16bf bb = load_b_frag(Kbase + (size_t)n0 * DDIM + kc * 32, DDIM, lane);
      s = wmma_bf16(a, bb, s);
    }

    // scale (1/sqrt(1024)) + causal mask
#pragma unroll
    for (int r = 0; r < 8; ++r) {
      const float v = s[r] * 0.03125f;
      const int mrow = m0 + r + rowoff;
      const int ncol = n0 + lh;
      s[r] = (ncol <= mrow) ? v : -1e30f;
    }

    // per-row max within wave (16-lane halves; wave32 shuffles)
#pragma unroll
    for (int r = 0; r < 8; ++r) {
      float v = s[r];
      v = fmaxf(v, __shfl_xor(v, 1, 32));
      v = fmaxf(v, __shfl_xor(v, 2, 32));
      v = fmaxf(v, __shfl_xor(v, 4, 32));
      v = fmaxf(v, __shfl_xor(v, 8, 32));
      if (lh == 0) wmaxs[w][r + rowoff] = v;
    }
    __syncthreads();

    if (tid < 16) {
      float mb = wmaxs[0][tid];
#pragma unroll
      for (int i = 1; i < 8; ++i) mb = fmaxf(mb, wmaxs[i][tid]);
      const float mo = rowmax[tid];
      const float mn = fmaxf(mo, mb);
      const float sc = __expf(mo - mn);
      rowscale[tid] = sc;
      rowmax[tid] = mn;
      rowsum[tid] *= sc;
    }
    __syncthreads();

    // P = exp(S - mnew); partial row sums; store P to LDS as bf16
#pragma unroll
    for (int r = 0; r < 8; ++r) {
      const float mn = rowmax[r + rowoff];
      const float p = __expf(s[r] - mn);
      Ps[(r + rowoff) * 128 + w * 16 + lh] = (__bf16)p;
      float t = p;
      t += __shfl_xor(t, 1, 32);
      t += __shfl_xor(t, 2, 32);
      t += __shfl_xor(t, 4, 32);
      t += __shfl_xor(t, 8, 32);
      if (lh == 0) wsums[w][r + rowoff] = t;
    }
    __syncthreads();

    if (tid < 16) {
      float tot = 0.0f;
#pragma unroll
      for (int i = 0; i < 8; ++i) tot += wsums[i][tid];
      rowsum[tid] += tot;
    }

    // rescale accumulators by exp(m_old - m_new)
    float scl[8];
#pragma unroll
    for (int r = 0; r < 8; ++r) scl[r] = rowscale[r + rowoff];
#pragma unroll
    for (int t = 0; t < 8; ++t)
#pragma unroll
      for (int r = 0; r < 8; ++r) acc[t][r] *= scl[r];

    // ---- O += P @ V  (wave w covers output cols [128w, 128w+128)) ----
#pragma unroll
    for (int kc = 0; kc < 4; ++kc) {
      v16bf a = load_a_frag(Ps + kc * 32, 128, lane);
#pragma unroll
      for (int t = 0; t < 8; ++t) {
        const __bf16* vp = Vbase + (size_t)(w * 128 + t * 16) * SLEN + k0 + kc * 32;
        v16bf bb = load_b_frag(vp, SLEN, lane);
        acc[t] = wmma_bf16(a, bb, acc[t]);
      }
    }
  }
  __syncthreads();

  float inv[8];
#pragma unroll
  for (int r = 0; r < 8; ++r) inv[r] = 1.0f / rowsum[r + rowoff];
#pragma unroll
  for (int t = 0; t < 8; ++t) {
    const int col = w * 128 + t * 16 + lh;
#pragma unroll
    for (int r = 0; r < 8; ++r) {
      const int mrow = m0 + r + rowoff;
      out[((size_t)b * SLEN + mrow) * DDIM + col] = acc[t][r] * inv[r];
    }
  }
}

// ---------------------------------------------------------------------------
extern "C" void kernel_launch(void* const* d_in, const int* in_sizes, int n_in,
                              void* d_out, int out_size, void* d_ws, size_t ws_size,
                              hipStream_t stream) {
  const float* x  = (const float*)d_in[0];
  const float* Wq = (const float*)d_in[1];
  const float* bq = (const float*)d_in[2];
  const float* Wk = (const float*)d_in[3];
  const float* bk = (const float*)d_in[4];
  const float* Wv = (const float*)d_in[5];
  const float* bv = (const float*)d_in[6];
  float* out = (float*)d_out;

  // workspace (bf16): Wqt|Wkt|Wvt (2MB ea) + Qb|Kb (32MB ea) + Vt (32MB) + xb (32MB)
  __bf16* Wqt = (__bf16*)d_ws;
  __bf16* Wkt = Wqt + (size_t)DDIM * DDIM;
  __bf16* Wvt = Wkt + (size_t)DDIM * DDIM;
  __bf16* Qb  = Wvt + (size_t)DDIM * DDIM;
  __bf16* Kb  = Qb + (size_t)BATCH * SLEN * DDIM;
  __bf16* Vt  = Kb + (size_t)BATCH * SLEN * DDIM;
  __bf16* xb  = Vt + (size_t)BATCH * SLEN * DDIM;

  xconv_kernel<<<dim3(((size_t)BATCH * SLEN * DDIM) / (256 * 8)), 256, 0, stream>>>(x, xb);
  wtrans_kernel<<<dim3((DDIM * DDIM) / 256, 1, 3), 256, 0, stream>>>(
      Wq, Wk, Wv, Wqt, Wkt, Wvt);
  proj_kernel<<<dim3(DDIM / 64, (BATCH * SLEN) / 64, 3), 256, 0, stream>>>(
      xb, Wqt, Wkt, Wvt, bq, bk, bv, Qb, Kb, Vt);
  attn_kernel<<<dim3(SLEN / 16, BATCH), 256, 0, stream>>>(Qb, Kb, Vt, out);

  (void)in_sizes; (void)n_in; (void)out_size; (void)ws_size;
}
